// DMC_77850577207773
// MI455X (gfx1250) — matrix operand order repaired and checked by reading, not compiled, not run on hardware
//
#include <hip/hip_runtime.h>
#include <hip/hip_bf16.h>

// ---------------------------------------------------------------------------
// Fused Mahalanobis-score loss for MI455X (gfx1250, wave32, WMMA bf16).
//   loss = sum_n 0.5 * min_k [ a_n - feat_n . G_k + d_k ]
//   a_n  = feat_n P feat_n^T,  G = (P + P^T) c^T,  d_k = c_k P c_k^T
//   feat = x @ W + b
// ---------------------------------------------------------------------------

typedef __attribute__((ext_vector_type(16))) __bf16 v16bf;
typedef __attribute__((ext_vector_type(8)))  __bf16 v8bf;
typedef __attribute__((ext_vector_type(8)))  float  v8f;

#define DIN   512
#define DD    256
#define KC    64
#define MTILE 128
#define LSTR  264   // LDS feat row stride in bf16 (256 + 8 pad, keeps 16B align)

// ---------------- prep kernels (tiny, run once per launch) ----------------

__global__ void zero_out_kernel(float* out) { out[0] = 0.0f; }

__global__ void __launch_bounds__(256) prep_wt_kernel(const float* __restrict__ W,
                                                      __bf16* __restrict__ WT) {
  int i = blockIdx.x * 256 + threadIdx.x;           // i over W [DIN x DD]
  if (i < DIN * DD) {
    int k = i / DD, n = i % DD;
    WT[(size_t)n * DIN + k] = (__bf16)W[i];         // WT[n][k] = W[k][n]
  }
}

__global__ void __launch_bounds__(256) prep_pt_kernel(const float* __restrict__ P,
                                                      __bf16* __restrict__ PT) {
  int i = blockIdx.x * 256 + threadIdx.x;           // i over P [DD x DD]
  if (i < DD * DD) {
    int k = i / DD, n = i % DD;
    PT[(size_t)n * DD + k] = (__bf16)P[i];          // PT[n][k] = P[k][n]
  }
}

// G = (P + P^T) c^T stored transposed: GT[kc][d]; also d_k = c_k P c_k^T.
__global__ void __launch_bounds__(256) prep_g_kernel(const float* __restrict__ C,
                                                     const float* __restrict__ P,
                                                     __bf16* __restrict__ GT,
                                                     float* __restrict__ dvec) {
  __shared__ float red;
  const int kc = blockIdx.x;     // cluster 0..63
  const int d  = threadIdx.x;    // feature dim 0..255
  if (d == 0) red = 0.f;
  __syncthreads();
  const float* crow = C + (size_t)kc * DD;
  float pd = 0.f, ptd = 0.f;
  for (int e = 0; e < DD; ++e) {
    float ce = crow[e];
    pd  += P[(size_t)d * DD + e] * ce;   // (P c^T)[d]
    ptd += P[(size_t)e * DD + d] * ce;   // (P^T c^T)[d]
  }
  GT[(size_t)kc * DD + d] = (__bf16)(pd + ptd);
  atomicAdd(&red, crow[d] * pd);         // c_k . (P c_k^T)
  __syncthreads();
  if (d == 0) dvec[kc] = red;
}

// ---------------- fused main kernel ----------------

__global__ void __launch_bounds__(256) maha_fused_kernel(
    const float* __restrict__ x, const float* __restrict__ bias,
    const __bf16* __restrict__ WT, const __bf16* __restrict__ PT,
    const __bf16* __restrict__ GT, const float* __restrict__ dvec,
    float* __restrict__ out) {
  __shared__ __bf16 featS[MTILE * LSTR];
  __shared__ float lsum;

  const int tid  = threadIdx.x;
  const int wave = tid >> 5;
  const int lane = tid & 31;
  const int l15  = lane & 15;
  const int h    = lane >> 4;        // K-half selector in WMMA fragments
  if (tid == 0) lsum = 0.f;

  const int m0 = blockIdx.x * MTILE + wave * 16;   // this wave's 16 rows

  v8f z8 = {0.f, 0.f, 0.f, 0.f, 0.f, 0.f, 0.f, 0.f};

  // -------- phase 1: feat = x @ W + b  (16 N-tiles per wave) --------
  v8f acc[16];
#pragma unroll
  for (int i = 0; i < 16; ++i) acc[i] = z8;

  const float* xrow = x + (size_t)(m0 + l15) * DIN;
#pragma unroll 4
  for (int kb = 0; kb < DIN; kb += 32) {
    // A fragment (16x32 bf16): lane holds K = kb + h*16 .. h*16+15
    const float* pa = xrow + kb + h * 16;
    float4 a0 = *(const float4*)(pa);
    float4 a1 = *(const float4*)(pa + 4);
    float4 a2 = *(const float4*)(pa + 8);
    float4 a3 = *(const float4*)(pa + 12);
    v16bf af;
    af[0]=(__bf16)a0.x;  af[1]=(__bf16)a0.y;  af[2]=(__bf16)a0.z;  af[3]=(__bf16)a0.w;
    af[4]=(__bf16)a1.x;  af[5]=(__bf16)a1.y;  af[6]=(__bf16)a1.z;  af[7]=(__bf16)a1.w;
    af[8]=(__bf16)a2.x;  af[9]=(__bf16)a2.y;  af[10]=(__bf16)a2.z; af[11]=(__bf16)a2.w;
    af[12]=(__bf16)a3.x; af[13]=(__bf16)a3.y; af[14]=(__bf16)a3.z; af[15]=(__bf16)a3.w;
#pragma unroll
    for (int nt = 0; nt < 16; ++nt) {
      // B fragment (32x16 bf16): lane = column N, contiguous 16 K values of WT row
      const __bf16* pb = WT + (size_t)(nt * 16 + l15) * DIN + kb + h * 16;
      v16bf bf = *(const v16bf*)pb;
      acc[nt] = __builtin_amdgcn_wmma_f32_16x16x32_bf16(
          false, af, false, bf, (short)0, acc[nt], false, false);
    }
  }

  // bias add + stash feat tile in LDS as bf16
#pragma unroll
  for (int nt = 0; nt < 16; ++nt) {
    float bv = bias[nt * 16 + l15];
#pragma unroll
    for (int r = 0; r < 8; ++r) {
      int rowl = wave * 16 + r + 8 * h;           // C/D layout: M = r + 8*h
      featS[rowl * LSTR + nt * 16 + l15] = (__bf16)(acc[nt][r] + bv);
    }
  }
  __syncthreads();

  // -------- phase 2: fP = feat @ P (16 tiles) and s = feat @ G (4 tiles) ----
  v8f facc[16], sacc[4];
#pragma unroll
  for (int i = 0; i < 16; ++i) facc[i] = z8;
#pragma unroll
  for (int i = 0; i < 4; ++i) sacc[i] = z8;

  const int arow = wave * 16 + l15;
#pragma unroll
  for (int kb = 0; kb < DD; kb += 32) {
    const __bf16* pl = &featS[arow * LSTR + kb + h * 16];
    v8bf lo = *(const v8bf*)pl;          // K = kb + h*16 .. +7     (ds_read_b128)
    v8bf hi = *(const v8bf*)(pl + 8);    // K = kb + h*16 + 8 .. +7
    v16bf af;
#pragma unroll
    for (int e = 0; e < 8; ++e) { af[e] = lo[e]; af[8 + e] = hi[e]; }
#pragma unroll
    for (int nt = 0; nt < 16; ++nt) {
      const __bf16* pb = PT + (size_t)(nt * 16 + l15) * DD + kb + h * 16;
      v16bf bf = *(const v16bf*)pb;
      facc[nt] = __builtin_amdgcn_wmma_f32_16x16x32_bf16(
          false, af, false, bf, (short)0, facc[nt], false, false);
    }
#pragma unroll
    for (int nt = 0; nt < 4; ++nt) {
      const __bf16* pb = GT + (size_t)(nt * 16 + l15) * DD + kb + h * 16;
      v16bf bf = *(const v16bf*)pb;
      sacc[nt] = __builtin_amdgcn_wmma_f32_16x16x32_bf16(
          false, af, false, bf, (short)0, sacc[nt], false, false);
    }
  }

  // a_m = fP_m . feat_m  (partial per lane over its 16 columns, then butterfly)
  float apart[8];
#pragma unroll
  for (int r = 0; r < 8; ++r) apart[r] = 0.f;
#pragma unroll
  for (int nt = 0; nt < 16; ++nt) {
#pragma unroll
    for (int r = 0; r < 8; ++r) {
      int rowl = wave * 16 + r + 8 * h;
      float fe = (float)featS[rowl * LSTR + nt * 16 + l15];
      apart[r] += facc[nt][r] * fe;
    }
  }
#pragma unroll
  for (int r = 0; r < 8; ++r) {
#pragma unroll
    for (int m = 1; m < 16; m <<= 1)
      apart[r] += __shfl_xor(apart[r], m, 32);   // stays within each K-half
  }

  // q_nk = a_n - s_nk + d_k ; row-wise min over 64 clusters
  float qmin[8];
#pragma unroll
  for (int r = 0; r < 8; ++r) qmin[r] = 3.4e38f;
#pragma unroll
  for (int nt = 0; nt < 4; ++nt) {
    float dv = dvec[nt * 16 + l15];
#pragma unroll
    for (int r = 0; r < 8; ++r) {
      float q = apart[r] - sacc[nt][r] + dv;
      qmin[r] = fminf(qmin[r], q);
    }
  }
#pragma unroll
  for (int r = 0; r < 8; ++r) {
#pragma unroll
    for (int m = 1; m < 16; m <<= 1)
      qmin[r] = fminf(qmin[r], __shfl_xor(qmin[r], m, 32));
  }

  if (l15 == 0) {   // one leader per (wave, half): 8 distinct rows each
    float s = 0.f;
#pragma unroll
    for (int r = 0; r < 8; ++r) s += 0.5f * qmin[r];
    atomicAdd(&lsum, s);
  }
  __syncthreads();
  if (tid == 0) atomicAdd(out, lsum);
}

// ---------------- host launcher ----------------

extern "C" void kernel_launch(void* const* d_in, const int* in_sizes, int n_in,
                              void* d_out, int out_size, void* d_ws, size_t ws_size,
                              hipStream_t stream) {
  const float* x    = (const float*)d_in[0];
  const float* W    = (const float*)d_in[1];
  const float* bias = (const float*)d_in[2];
  const float* C    = (const float*)d_in[3];
  const float* P    = (const float*)d_in[4];
  float* out        = (float*)d_out;

  char* ws = (char*)d_ws;
  __bf16* WT  = (__bf16*)(ws + 0);        // 256x512 bf16 = 256 KB
  __bf16* PT  = (__bf16*)(ws + 262144);   // 256x256 bf16 = 128 KB
  __bf16* GT  = (__bf16*)(ws + 393216);   //  64x256 bf16 =  32 KB
  float*  dvec = (float*)(ws + 425984);   //  64 f32

  const int N = in_sizes[0] / DIN;

  hipLaunchKernelGGL(zero_out_kernel, dim3(1), dim3(1), 0, stream, out);
  hipLaunchKernelGGL(prep_wt_kernel, dim3((DIN * DD + 255) / 256), dim3(256), 0, stream, W, WT);
  hipLaunchKernelGGL(prep_pt_kernel, dim3((DD * DD + 255) / 256), dim3(256), 0, stream, P, PT);
  hipLaunchKernelGGL(prep_g_kernel, dim3(KC), dim3(DD), 0, stream, C, P, GT, dvec);
  hipLaunchKernelGGL(maha_fused_kernel, dim3(N / MTILE), dim3(256), 0, stream,
                     x, bias, WT, PT, GT, dvec, out);
}
